// MultiHeadAttention_65360812310777
// MI455X (gfx1250) — matrix operand order repaired and checked by reading, not compiled, not run on hardware
//
#include <hip/hip_runtime.h>

typedef __attribute__((ext_vector_type(2)))  float     v2f;
typedef __attribute__((ext_vector_type(8)))  float     v8f;
typedef __attribute__((ext_vector_type(16))) _Float16  v16h;
typedef int v4i __attribute__((vector_size(16)));   // matches async-LDS builtin param

#define AS1 __attribute__((address_space(1)))
#define AS3 __attribute__((address_space(3)))

#define A_STRIDE 132   // padded row stride: bank = (4*m + c) % 64, conflict-free;
                       // 132*4 = 528 bytes = 33*16 -> every float4 stays 16B-aligned

// ---------------------------------------------------------------------------
// Projection GEMM: O[16-row tile] = X_tile(16x128) @ W(128x128) + bias
// grid.x = row tiles, grid.y = {0:Q, 1:K, 2:V}; 256 threads = 8 waves,
// wave w computes the 16x16 output tile at columns [16w, 16w+16).
// ---------------------------------------------------------------------------
__global__ __launch_bounds__(256) void gatt_proj(
    const float* __restrict__ h_src, const float* __restrict__ h_dst,
    const float* __restrict__ Qw, const float* __restrict__ Qb,
    const float* __restrict__ Kw, const float* __restrict__ Kb,
    const float* __restrict__ Vw, const float* __restrict__ Vb,
    float* __restrict__ Qo, float* __restrict__ Ko, float* __restrict__ Vo,
    int N)
{
    __shared__ float As[16][A_STRIDE];

    const int tid   = threadIdx.x;
    const int wave  = tid >> 5;
    const int lane  = tid & 31;
    const int mtile = blockIdx.x;
    const int which = blockIdx.y;

    const float* X  = (which == 0) ? h_dst : h_src;
    const float* W  = (which == 0) ? Qw : (which == 1) ? Kw : Vw;
    const float* Bv = (which == 0) ? Qb : (which == 1) ? Kb : Vb;
    float*       O  = (which == 0) ? Qo : (which == 1) ? Ko : Vo;

    // ---- Stage the 16x128 A tile into LDS ---------------------------------
#if __has_builtin(__builtin_amdgcn_global_load_async_to_lds_b128)
    // CDNA5 async copy engine: global -> LDS directly, tracked by ASYNCcnt.
    // 512 B128 transfers (16 rows x 32 float4), 2 per thread.
    for (int t = tid; t < 512; t += 256) {
        const int r    = t >> 5;        // row in tile
        const int c4   = t & 31;        // float4 within row
        const int grow = mtile * 16 + r;
        if (grow < N) {
            const float* gsrc = X + (size_t)grow * 128 + c4 * 4;
            float*       ldst = &As[r][c4 * 4];
            __builtin_amdgcn_global_load_async_to_lds_b128(
                (AS1 v4i*)gsrc, (AS3 v4i*)ldst, 0, 0);
        } else {
            float* ldst = &As[r][c4 * 4];
            ldst[0] = 0.f; ldst[1] = 0.f; ldst[2] = 0.f; ldst[3] = 0.f;
        }
    }
#if __has_builtin(__builtin_amdgcn_s_wait_asynccnt)
    __builtin_amdgcn_s_wait_asynccnt(0);
#else
    asm volatile("s_wait_asynccnt 0" ::: "memory");
#endif
#else
    // Fallback: synchronous staging through VGPRs.
    for (int idx = tid; idx < 16 * 128; idx += 256) {
        const int r = idx >> 7, c = idx & 127;
        const int grow = mtile * 16 + r;
        As[r][c] = (grow < N) ? X[(size_t)grow * 128 + c] : 0.0f;
    }
#endif
    __syncthreads();

    // WMMA fragment coordinates (wave32).
    const int m     = lane & 15;          // A row / D column within the tile
    const int khalf = lane >> 4;          // 0: lanes 0-15, 1: lanes 16-31
    const int n     = (lane & 15) + wave * 16;  // B/D column

    v8f acc = {};

#if __has_builtin(__builtin_amdgcn_wmma_f32_16x16x4_f32)
    // Exact fp32 path: 32 k-steps of 16x16x4.
    // A frag: lane holds A[m][ka], A[m][ka+1] with ka = k + 2*khalf.
    // B frag: lane holds W[ka][n], W[ka+1][n].
    #pragma unroll 4
    for (int k = 0; k < 128; k += 4) {
        const int ka = k + khalf * 2;
        v2f a, b;
        a.x = As[m][ka];
        a.y = As[m][ka + 1];
        b.x = W[(size_t)ka       * 128 + n];
        b.y = W[(size_t)(ka + 1) * 128 + n];
        acc = __builtin_amdgcn_wmma_f32_16x16x4_f32(
            false, a, false, b, (short)0, acc, false, false);
    }
#else
    // Fallback: f16 inputs, f32 accumulate (probe-confirmed builtin).
    for (int k0 = 0; k0 < 128; k0 += 32) {
        v16h a, b;
        #pragma unroll
        for (int j = 0; j < 16; ++j) {
            const int ka = khalf * 16 + j;
            const int kb = khalf * 16 + j;
            a[j] = (_Float16)As[m][k0 + ka];
            b[j] = (_Float16)W[(size_t)(k0 + kb) * 128 + n];
        }
        acc = __builtin_amdgcn_wmma_f32_16x16x32_f16(
            false, a, false, b, (short)0, acc, false, false);
    }
#endif

    // D layout: VGPR r -> M = r + 8*khalf, N = n.
    const float bias  = Bv[n];
    const int   mbase = mtile * 16 + khalf * 8;
    #pragma unroll
    for (int r = 0; r < 8; ++r) {
        const int mm = mbase + r;
        if (mm < N) O[(size_t)mm * 128 + n] = acc[r] + bias;
    }
}

// ---------------------------------------------------------------------------
// Edge phase: one edge per 16-lane half-wave. Lane i owns dim i of each head.
// score[h] = exp(clamp(dot16(K[s,h],Q[d,h]) * 0.25, -5, 5)); atomic scatter.
// Q/K/V (76.8 MB) fit in the 192 MB L2, so gathers + atomics resolve in L2.
// ---------------------------------------------------------------------------
__global__ __launch_bounds__(256) void gatt_edge(
    const float* __restrict__ Q, const float* __restrict__ K,
    const float* __restrict__ V,
    const int* __restrict__ src, const int* __restrict__ dst,
    float* __restrict__ wV, float* __restrict__ z, int E)
{
    const int half = threadIdx.x >> 4;          // 16 edges per 256-thread block
    const int i    = threadIdx.x & 15;          // dim within head
    const int e    = blockIdx.x * 16 + half;
    if (e >= E) return;

    const int s = src[e];
    const int d = dst[e];
    const float* Ks  = K  + (size_t)s * 128;
    const float* Qd  = Q  + (size_t)d * 128;
    const float* Vs  = V  + (size_t)s * 128;
    float*       wVd = wV + (size_t)d * 128;
    float*       zd  = z  + (size_t)d * 8;

    __builtin_prefetch(Vs, 0, 0);               // global_prefetch_b8
    __builtin_prefetch(Vs + 64, 0, 0);

    #pragma unroll
    for (int h = 0; h < 8; ++h) {
        const int o = h * 16 + i;
        float p = Ks[o] * Qd[o];
        // reduce over the 16 lanes of this half-wave (wave32)
        p += __shfl_xor(p, 1, 16);
        p += __shfl_xor(p, 2, 16);
        p += __shfl_xor(p, 4, 16);
        p += __shfl_xor(p, 8, 16);
        float sc = p * 0.25f;                   // 1/sqrt(D), D=16
        sc = fminf(5.0f, fmaxf(-5.0f, sc));
        const float w = expf(sc);
        atomicAdd(&wVd[o], w * Vs[o]);
        if (i == 0) atomicAdd(&zd[h], w);
    }
}

// ---------------------------------------------------------------------------
__global__ void gatt_zero(float* __restrict__ p, size_t count)
{
    size_t idx = (size_t)blockIdx.x * blockDim.x + threadIdx.x;
    const size_t stride = (size_t)gridDim.x * blockDim.x;
    for (; idx < count; idx += stride) p[idx] = 0.0f;
}

__global__ void gatt_norm(const float* __restrict__ wV,
                          const float* __restrict__ z,
                          float* __restrict__ out, int total)
{
    const int idx = blockIdx.x * blockDim.x + threadIdx.x;
    if (idx >= total) return;
    const int node = idx >> 7;          // /128
    const int head = (idx >> 4) & 7;    // /16 % 8
    out[idx] = wV[idx] / z[node * 8 + head];
}

// ---------------------------------------------------------------------------
extern "C" void kernel_launch(void* const* d_in, const int* in_sizes, int n_in,
                              void* d_out, int out_size, void* d_ws, size_t ws_size,
                              hipStream_t stream)
{
    const float* h_src = (const float*)d_in[0];
    const float* h_dst = (const float*)d_in[1];
    const int*   src   = (const int*)  d_in[2];
    const int*   dst   = (const int*)  d_in[3];
    const float* Qw    = (const float*)d_in[4];
    const float* Qb    = (const float*)d_in[5];
    const float* Kw    = (const float*)d_in[6];
    const float* Kb    = (const float*)d_in[7];
    const float* Vw    = (const float*)d_in[8];
    const float* Vb    = (const float*)d_in[9];

    const int N = in_sizes[0] / 128;
    const int E = in_sizes[2];

    // Workspace layout: Q | K | V | wV | z   (floats)
    float*       ws   = (float*)d_ws;
    const size_t n128 = (size_t)N * 128;
    float* Qp  = ws;
    float* Kp  = ws + n128;
    float* Vp  = ws + 2 * n128;
    float* wVp = ws + 3 * n128;
    float* zp  = ws + 4 * n128;

    // 1) zero accumulators (wV then z are contiguous)
    const size_t zcount = n128 + (size_t)N * 8;
    gatt_zero<<<2048, 256, 0, stream>>>(wVp, zcount);

    // 2) Q/K/V projections via WMMA
    const int mtiles = (N + 15) / 16;
    gatt_proj<<<dim3(mtiles, 3), 256, 0, stream>>>(
        h_src, h_dst, Qw, Qb, Kw, Kb, Vw, Vb, Qp, Kp, Vp, N);

    // 3) edge scores + atomic scatter-sum
    gatt_edge<<<(E + 15) / 16, 256, 0, stream>>>(Qp, Kp, Vp, src, dst, wVp, zp, E);

    // 4) normalize
    const int total = N * 128;
    gatt_norm<<<(total + 255) / 256, 256, 0, stream>>>(wVp, zp, (float*)d_out, total);
}
